// GraphConvolution1d_86346022518990
// MI455X (gfx1250) — matrix operand order repaired and checked by reading, not compiled, not run on hardware
//
#include <hip/hip_runtime.h>
#include <hip/hip_bf16.h>

typedef __attribute__((ext_vector_type(16))) __bf16 v16bf;
typedef __attribute__((ext_vector_type(8)))  float  v8f;

#define NROWS 16384
#define FDIM  1024
#define NLAYR 5
#define BK    32
#define BM    256
#define BN    128
#define LDS_STRIDE 40   // halfs; 80B row stride -> 16B aligned, conflict-free b128 reads

// ---------------------------------------------------------------------------
// fp32 -> bf16 bulk convert (4 elements / thread)
// ---------------------------------------------------------------------------
__global__ __launch_bounds__(256) void cvt_f32_bf16(const float* __restrict__ x,
                                                    __bf16* __restrict__ o,
                                                    long n) {
  long i = (long)blockIdx.x * blockDim.x + threadIdx.x;
  long base = i * 4;
  if (base >= n) return;
  float4 f = *(const float4*)(x + base);
  union { __bf16 h[4]; uint2 u; } p;
  p.h[0] = (__bf16)f.x; p.h[1] = (__bf16)f.y;
  p.h[2] = (__bf16)f.z; p.h[3] = (__bf16)f.w;
  *(uint2*)(o + base) = p.u;
}

// ---------------------------------------------------------------------------
// Bt[n][k] = bf16( W[k][n] * adj[k][n] )  -- LDS tiled transpose, 32x32 tiles
// ---------------------------------------------------------------------------
__global__ __launch_bounds__(256) void prep_bt(const float* __restrict__ W,
                                               const float* __restrict__ adj,
                                               __bf16* __restrict__ Bt) {
  __shared__ float tile[32][33];
  const int n0 = blockIdx.x * 32;
  const int k0 = blockIdx.y * 32;
  const int tx = threadIdx.x;   // 0..31
  const int ty = threadIdx.y;   // 0..7
  for (int j = 0; j < 4; ++j) {
    const int kl = ty + j * 8;
    const size_t idx = (size_t)(k0 + kl) * FDIM + n0 + tx;
    tile[kl][tx] = W[idx] * adj[idx];
  }
  __syncthreads();
  for (int j = 0; j < 4; ++j) {
    const int nl = ty + j * 8;
    Bt[(size_t)(n0 + nl) * FDIM + k0 + tx] = (__bf16)tile[tx][nl];
  }
}

// ---------------------------------------------------------------------------
// Y = relu(Hb @ Bt^T + bias) + Hb      (fp32 out; A,B bf16; fp32 accumulate)
// Block tile 256(M) x 128(N), K-step 32, 8 waves in a 4x2 grid,
// each wave: 64x64 = 4x4 tiles of 16x16 -> 16 v8f accumulators (128 VGPRs).
// Per wave/K-step: 16 ds_load_b128 feed 16 v_wmma (1.0 LDS reads / WMMA).
// ---------------------------------------------------------------------------
__global__ __launch_bounds__(256) void gemm_relu_res(
    const __bf16* __restrict__ Hb, const __bf16* __restrict__ Bt,
    const float* __restrict__ bias, float* __restrict__ Y) {
  __shared__ __bf16 As[2][BM * LDS_STRIDE];   // 2 * 20 KB
  __shared__ __bf16 Bs[2][BN * LDS_STRIDE];   // 2 * 10 KB

  const int tid  = threadIdx.x;
  const int lane = tid & 31;
  const int ln   = lane & 15;
  const int hi   = lane >> 4;
  const int wave = tid >> 5;
  const int wy   = wave >> 1;   // 0..3 -> M offset wy*64
  const int wx   = wave & 1;    // 0..1 -> N offset wx*64

  const int m0 = blockIdx.x * BM;
  const int n0 = blockIdx.y * BN;

  // global->LDS mapping: each thread moves 4 A-rows + 2 B-rows x 8 halfs
  const int r0  = tid >> 2;     // 0..63
  const int seg = tid & 3;      // k sub-offset = seg*8

  const __bf16* Ag = Hb + (size_t)m0 * FDIM + seg * 8;
  const __bf16* Bg = Bt + (size_t)n0 * FDIM + seg * 8;

  v8f acc[4][4];
  for (int mt = 0; mt < 4; ++mt)
    for (int nt = 0; nt < 4; ++nt)
      for (int r = 0; r < 8; ++r)
        acc[mt][nt][r] = 0.f;

  // preload k-tile 0
  {
    uint4 a0 = *(const uint4*)(Ag + (size_t)(r0      ) * FDIM);
    uint4 a1 = *(const uint4*)(Ag + (size_t)(r0 +  64) * FDIM);
    uint4 a2 = *(const uint4*)(Ag + (size_t)(r0 + 128) * FDIM);
    uint4 a3 = *(const uint4*)(Ag + (size_t)(r0 + 192) * FDIM);
    uint4 b0 = *(const uint4*)(Bg + (size_t)(r0      ) * FDIM);
    uint4 b1 = *(const uint4*)(Bg + (size_t)(r0 +  64) * FDIM);
    *(uint4*)&As[0][(r0      ) * LDS_STRIDE + seg * 8] = a0;
    *(uint4*)&As[0][(r0 +  64) * LDS_STRIDE + seg * 8] = a1;
    *(uint4*)&As[0][(r0 + 128) * LDS_STRIDE + seg * 8] = a2;
    *(uint4*)&As[0][(r0 + 192) * LDS_STRIDE + seg * 8] = a3;
    *(uint4*)&Bs[0][(r0      ) * LDS_STRIDE + seg * 8] = b0;
    *(uint4*)&Bs[0][(r0 +  64) * LDS_STRIDE + seg * 8] = b1;
  }
  __syncthreads();

  const int KT = FDIM / BK;   // 32
  for (int kt = 0; kt < KT; ++kt) {
    const int cur = kt & 1;
    const bool more = (kt + 1 < KT);
    uint4 a0, a1, a2, a3, b0, b1;
    if (more) {
      const size_t ko = (size_t)(kt + 1) * BK;
      a0 = *(const uint4*)(Ag + (size_t)(r0      ) * FDIM + ko);
      a1 = *(const uint4*)(Ag + (size_t)(r0 +  64) * FDIM + ko);
      a2 = *(const uint4*)(Ag + (size_t)(r0 + 128) * FDIM + ko);
      a3 = *(const uint4*)(Ag + (size_t)(r0 + 192) * FDIM + ko);
      b0 = *(const uint4*)(Bg + (size_t)(r0      ) * FDIM + ko);
      b1 = *(const uint4*)(Bg + (size_t)(r0 +  64) * FDIM + ko);
    }
    if (kt + 2 < KT) {
      const size_t ko2 = (size_t)(kt + 2) * BK;
      __builtin_prefetch(Ag + (size_t)r0 * FDIM + ko2, 0, 3);
      __builtin_prefetch(Bg + (size_t)r0 * FDIM + ko2, 0, 3);
    }

    // A fragments: ISA 16-bit A 16x32 layout.
    // lanes 0-15 hold K {0..7, 16..23}; lanes 16-31 hold K {8..15, 24..31}
    union { v16bf v; uint4 u[2]; } af[4], bfr[4];
    for (int mt = 0; mt < 4; ++mt) {
      const int row = wy * 64 + mt * 16 + ln;
      af[mt].u[0] = *(const uint4*)&As[cur][row * LDS_STRIDE + hi * 8];
      af[mt].u[1] = *(const uint4*)&As[cur][row * LDS_STRIDE + hi * 8 + 16];
    }
    // B fragments: lanes 0-15 hold K {0..7, 16..23}; lanes 16-31 hold K {8..15, 24..31}
    for (int nt = 0; nt < 4; ++nt) {
      const int col = wx * 64 + nt * 16 + ln;
      bfr[nt].u[0] = *(const uint4*)&Bs[cur][col * LDS_STRIDE + hi * 8];
      bfr[nt].u[1] = *(const uint4*)&Bs[cur][col * LDS_STRIDE + hi * 8 + 16];
    }

    for (int mt = 0; mt < 4; ++mt)
      for (int nt = 0; nt < 4; ++nt)
        acc[mt][nt] = __builtin_amdgcn_wmma_f32_16x16x32_bf16(
            false, af[mt].v, false, bfr[nt].v, (short)0, acc[mt][nt],
            false, false);

    if (more) {
      const int nxt = cur ^ 1;
      *(uint4*)&As[nxt][(r0      ) * LDS_STRIDE + seg * 8] = a0;
      *(uint4*)&As[nxt][(r0 +  64) * LDS_STRIDE + seg * 8] = a1;
      *(uint4*)&As[nxt][(r0 + 128) * LDS_STRIDE + seg * 8] = a2;
      *(uint4*)&As[nxt][(r0 + 192) * LDS_STRIDE + seg * 8] = a3;
      *(uint4*)&Bs[nxt][(r0      ) * LDS_STRIDE + seg * 8] = b0;
      *(uint4*)&Bs[nxt][(r0 +  64) * LDS_STRIDE + seg * 8] = b1;
    }
    __syncthreads();
  }

  // fused epilogue: bias + relu + residual (bf16 H), fp32 store
  float bv[4];
  for (int nt = 0; nt < 4; ++nt)
    bv[nt] = bias[n0 + wx * 64 + nt * 16 + ln];

  for (int mt = 0; mt < 4; ++mt) {
    for (int nt = 0; nt < 4; ++nt) {
      const int col = n0 + wx * 64 + nt * 16 + ln;
      for (int r = 0; r < 8; ++r) {
        const int row = m0 + wy * 64 + mt * 16 + hi * 8 + r;
        float v = acc[mt][nt][r] + bv[nt];
        v = v > 0.f ? v : 0.f;
        v += (float)Hb[(size_t)row * FDIM + col];
        Y[(size_t)row * FDIM + col] = v;
      }
    }
  }
}

// ---------------------------------------------------------------------------
// Row LayerNorm: 1 block / row, 256 threads x 4 elements.
// Writes bf16 H for next layer, or fp32 d_out on the final layer.
// ---------------------------------------------------------------------------
__global__ __launch_bounds__(256) void layernorm_rows(const float* __restrict__ Y,
                                                      float* __restrict__ outF,
                                                      __bf16* __restrict__ outB) {
  const int row = blockIdx.x;
  const float* yr = Y + (size_t)row * FDIM;
  const int t = threadIdx.x;

  float v[4];
  float s = 0.f, ss = 0.f;
  for (int j = 0; j < 4; ++j) {
    v[j] = yr[t + j * 256];
    s  += v[j];
    ss += v[j] * v[j];
  }
  for (int off = 16; off > 0; off >>= 1) {
    s  += __shfl_down(s, off, 32);
    ss += __shfl_down(ss, off, 32);
  }
  __shared__ float red[2][8];
  __shared__ float stats[2];
  const int wv = t >> 5, lane = t & 31;
  if (lane == 0) { red[0][wv] = s; red[1][wv] = ss; }
  __syncthreads();
  if (t == 0) {
    float S = 0.f, SS = 0.f;
    for (int w = 0; w < 8; ++w) { S += red[0][w]; SS += red[1][w]; }
    const float mu  = S * (1.f / FDIM);
    const float var = SS * (1.f / FDIM) - mu * mu;
    stats[0] = mu;
    stats[1] = rsqrtf(var + 1e-5f);
  }
  __syncthreads();
  const float mu = stats[0], inv = stats[1];
  for (int j = 0; j < 4; ++j) {
    const float o = (v[j] - mu) * inv;
    const size_t idx = (size_t)row * FDIM + t + j * 256;
    if (outF) outF[idx] = o;
    if (outB) outB[idx] = (__bf16)o;
  }
}

// ---------------------------------------------------------------------------
extern "C" void kernel_launch(void* const* d_in, const int* in_sizes, int n_in,
                              void* d_out, int out_size, void* d_ws, size_t ws_size,
                              hipStream_t stream) {
  const float* x    = (const float*)d_in[0];   // (16384, 1024) fp32
  const float* adj  = (const float*)d_in[1];   // (1024, 1024)  fp32
  const float* Wts  = (const float*)d_in[2];   // (5, 1024, 1024) fp32
  const float* bias = (const float*)d_in[3];   // (5, 1024) fp32
  float* out = (float*)d_out;                  // (16384, 1024) fp32

  char* ws = (char*)d_ws;
  float*  Y  = (float*)ws;                                               // 64 MB
  __bf16* Hb = (__bf16*)(ws + (size_t)NROWS * FDIM * 4);                 // 32 MB
  __bf16* Bt = (__bf16*)(ws + (size_t)NROWS * FDIM * 4
                            + (size_t)NROWS * FDIM * 2);                 //  2 MB

  const long total = (long)NROWS * FDIM;
  cvt_f32_bf16<<<(int)((total / 4 + 255) / 256), 256, 0, stream>>>(x, Hb, total);

  for (int l = 0; l < NLAYR; ++l) {
    prep_bt<<<dim3(FDIM / 32, FDIM / 32), dim3(32, 8), 0, stream>>>(
        Wts + (size_t)l * FDIM * FDIM, adj, Bt);
    gemm_relu_res<<<dim3(NROWS / BM, FDIM / BN), 256, 0, stream>>>(
        Hb, Bt, bias + (size_t)l * FDIM, Y);
    const bool last = (l == NLAYR - 1);
    layernorm_rows<<<NROWS, 256, 0, stream>>>(
        Y, last ? out : nullptr, last ? nullptr : Hb);
  }
}
